// Block_88158498717701
// MI455X (gfx1250) — matrix operand order repaired and checked by reading, not compiled, not run on hardware
//
#include <hip/hip_runtime.h>

// ---------- types ----------
typedef __attribute__((ext_vector_type(16))) __bf16       v16bf;
typedef __attribute__((ext_vector_type(8)))  float        v8f;
typedef __attribute__((ext_vector_type(4)))  unsigned int v4u;
typedef __attribute__((ext_vector_type(8)))  int          v8i;
typedef __attribute__((ext_vector_type(4)))  int          v4i;

union Frag32 { v4u u[2]; v16bf v; };   // 32 bytes = one 16x32 bf16 A/B fragment per lane

#if defined(__gfx1250__) && __has_builtin(__builtin_amdgcn_tensor_load_to_lds) && \
    __has_builtin(__builtin_amdgcn_s_wait_tensorcnt)
#define USE_TDM 1
#else
#define USE_TDM 0
#endif

__device__ __forceinline__ unsigned short f2bf(float f) {
  unsigned int u = __float_as_uint(f);
  u += 0x7FFFu + ((u >> 16) & 1u);     // round-to-nearest-even
  return (unsigned short)(u >> 16);
}

__device__ __forceinline__ v8f wmma_bf16(const Frag32& a, const Frag32& b, v8f c) {
  return __builtin_amdgcn_wmma_f32_16x16x32_bf16(false, a.v, false, b.v, (short)0, c, false, false);
}

#if USE_TDM
// TDM: load a 2D tile of 128 rows x 32 bf16 (contiguous) from row-major global
// (row stride ldElems) into LDS at byte offset ldsOff, padding each 64B row to 96B
// (pad_interval code 3 = 16 DWORDs, pad_amount code 7 = 8 DWORDs -> 48-short rows).
__device__ __forceinline__ void tdm_load_tile_128x32(unsigned ldsOff,
                                                     const unsigned short* gptr,
                                                     unsigned ldElems) {
  unsigned long long ga = (unsigned long long)(uintptr_t)gptr;
  v4u g0;
  g0[0] = 1u;                                   // count=1 valid descriptor
  g0[1] = ldsOff;                               // lds_addr (bytes)
  g0[2] = (unsigned)ga;                         // global_addr[31:0]
  g0[3] = (unsigned)((ga >> 32) & 0x1FFFFFFu) | (2u << 30);  // addr[56:32] | type=2
  v8i g1;
  g1[0] = (int)((1u << 16) | (1u << 20) | (3u << 22) | (7u << 25)); // 2B elems, pad en
  g1[1] = (int)((ldElems & 0xFFFFu) << 16);                 // tensor_dim0[15:0]
  g1[2] = (int)(((ldElems >> 16) & 0xFFFFu));               // tensor_dim0[31:16], dim1 lo=0
  g1[3] = (int)(0x10u | (32u << 16));                       // dim1=0x100000, tile_dim0=32
  g1[4] = (int)128u;                                        // tile_dim1=128, tile_dim2=0
  g1[5] = (int)ldElems;                                     // tensor_dim0_stride[31:0]
  g1[6] = 0;                                                // stride hi / dim1 stride
  g1[7] = 0;
  v4i z4 = {0, 0, 0, 0};
  v8i z8 = {0, 0, 0, 0, 0, 0, 0, 0};
  __builtin_amdgcn_tensor_load_to_lds(g0, g1, z4, z4, z8, 0);
}
#endif

// ---------- fp32 -> bf16 conversion ----------
__global__ __launch_bounds__(256) void cvt_bf16_kernel(const float* __restrict__ in,
                                                       unsigned short* __restrict__ out, int n) {
  int i = blockIdx.x * 256 + threadIdx.x;
  int stride = gridDim.x * 256;
  for (; i < n; i += stride) out[i] = f2bf(in[i]);
}

// fp32 [R][C] -> bf16 transposed [C][R]
__global__ __launch_bounds__(256) void cvt_bf16_t_kernel(const float* __restrict__ in,
                                                         unsigned short* __restrict__ out,
                                                         int R, int C) {
  int i = blockIdx.x * 256 + threadIdx.x;
  int n = R * C;
  int stride = gridDim.x * 256;
  for (; i < n; i += stride) {
    int r = i / C, c = i - r * C;
    out[(size_t)c * R + r] = f2bf(in[i]);
  }
}

// ---------- tiled bf16 WMMA GEMM:  C[M,N] = A[M,K] @ (BT[N,K])^T ----------
// Block tile 128x128, K-step 32, double-buffered TDM tile DMA into LDS.
// 8 waves: wave (wm,wn) computes 64x32 (4x2 16x16 tiles).
// MODE 0: QKV scatter -> q[BH,T,64], k[BH,T,64], vT[BH,64,T] (bf16, +bias)
// MODE 1: bf16 out, relu(acc+bias)
// MODE 2: f32 out, acc+bias
template <int MODE>
__global__ __launch_bounds__(256) void gemm_kernel(
    const unsigned short* __restrict__ Ag, const unsigned short* __restrict__ BTg,
    const float* __restrict__ bias, void* __restrict__ C0, void* __restrict__ C1,
    void* __restrict__ C2, int M, int N, int K)
{
  // [buf][128 rows][32 elems padded to 48]
  __shared__ __align__(16) unsigned short As[2 * 128 * 48];
  __shared__ __align__(16) unsigned short Bs[2 * 128 * 48];
  const int t    = threadIdx.x;
  const int lane = t & 31, w = t >> 5;
  const int hf   = lane >> 4, l16 = lane & 15;
  const int wm   = w & 1,  wn = w >> 1;
  const int bm   = blockIdx.y, bn = blockIdx.x;
  const unsigned short* Abase = Ag  + (size_t)(bm * 128) * K;
  const unsigned short* Bbase = BTg + (size_t)(bn * 128) * K;

  const v8f zero8 = {0.f,0.f,0.f,0.f,0.f,0.f,0.f,0.f};
  v8f acc[4][2];
#pragma unroll
  for (int i = 0; i < 4; ++i)
#pragma unroll
    for (int j = 0; j < 2; ++j) acc[i][j] = zero8;

  const int nsteps = K >> 5;
#if USE_TDM
  const unsigned ldsA = (unsigned)(uintptr_t)&As[0];
  const unsigned ldsB = (unsigned)(uintptr_t)&Bs[0];
  if (w == 0) {  // one wave drives the TDM (EXEC-independent, per-wave TENSORcnt)
    tdm_load_tile_128x32(ldsA, Abase, (unsigned)K);
    tdm_load_tile_128x32(ldsB, Bbase, (unsigned)K);
  }
#endif
  for (int s = 0; s < nsteps; ++s) {
    const int cur = s & 1;
    const int k0 = s << 5;
#if USE_TDM
    if (w == 0) {
      if (s + 1 < nsteps) {
        const int nb = (s + 1) & 1;
        tdm_load_tile_128x32(ldsA + nb * (128 * 48 * 2), Abase + k0 + 32, (unsigned)K);
        tdm_load_tile_128x32(ldsB + nb * (128 * 48 * 2), Bbase + k0 + 32, (unsigned)K);
        __builtin_amdgcn_s_wait_tensorcnt(2);   // current pair of tiles landed
      } else {
        __builtin_amdgcn_s_wait_tensorcnt(0);
      }
    }
    __syncthreads();
#else
    // fallback: synchronous staging (both tiles are 128 rows x 32 contiguous)
#pragma unroll
    for (int i = 0; i < 2; ++i) {
      int c = t + i * 256;
      int r = c >> 2, c8 = (c & 3) << 3;
      *(v4u*)(&As[cur * 128 * 48 + r * 48 + c8]) = *(const v4u*)(Abase + (size_t)r * K + k0 + c8);
      *(v4u*)(&Bs[cur * 128 * 48 + r * 48 + c8]) = *(const v4u*)(Bbase + (size_t)r * K + k0 + c8);
    }
    __syncthreads();
#endif
    Frag32 af[4], bfr[2];
#pragma unroll
    for (int mt = 0; mt < 4; ++mt) {
      const unsigned short* p = &As[cur * 128 * 48 + (wm * 64 + mt * 16 + l16) * 48 + hf * 8];
      af[mt].u[0] = *(const v4u*)p;          // K: half?8..15:0..7
      af[mt].u[1] = *(const v4u*)(p + 16);   // K: half?24..31:16..23
    }
#pragma unroll
    for (int nt = 0; nt < 2; ++nt) {
      const unsigned short* p = &Bs[cur * 128 * 48 + (wn * 32 + nt * 16 + l16) * 48 + hf * 8];
      bfr[nt].u[0] = *(const v4u*)p;         // K: half?8..15:0..7
      bfr[nt].u[1] = *(const v4u*)(p + 16);  // K: half?24..31:16..23
    }
#pragma unroll
    for (int mt = 0; mt < 4; ++mt)
#pragma unroll
      for (int nt = 0; nt < 2; ++nt)
        acc[mt][nt] = wmma_bf16(af[mt], bfr[nt], acc[mt][nt]);
    __syncthreads();  // all reads of buf `cur` done before it is overwritten
  }

  // epilogue: C layout: row = jj + 8*half, col = lane%16
#pragma unroll
  for (int mt = 0; mt < 4; ++mt) {
#pragma unroll
    for (int nt = 0; nt < 2; ++nt) {
      int j  = bn * 128 + wn * 32 + nt * 16 + l16;
      float bj = bias ? bias[j] : 0.f;
#pragma unroll
      for (int jj = 0; jj < 8; ++jj) {
        int i = bm * 128 + wm * 64 + mt * 16 + hf * 8 + jj;
        float v = acc[mt][nt][jj] + bj;
        if (MODE == 0) {
          int kind = j >> 10;
          int jm = j & 1023;
          int hh = jm >> 6, dd = jm & 63;
          int bb = i >> 11, tt = i & 2047;
          int bh = bb * 16 + hh;
          unsigned short bvs = f2bf(v);
          if (kind == 0)      ((unsigned short*)C0)[((size_t)bh * 2048 + tt) * 64 + dd] = bvs;
          else if (kind == 1) ((unsigned short*)C1)[((size_t)bh * 2048 + tt) * 64 + dd] = bvs;
          else                ((unsigned short*)C2)[((size_t)bh * 64 + dd) * 2048 + tt] = bvs;
        } else if (MODE == 1) {
          v = v > 0.f ? v : 0.f;
          ((unsigned short*)C0)[(size_t)i * N + j] = f2bf(v);
        } else {
          ((float*)C0)[(size_t)i * N + j] = v;
        }
      }
    }
  }
}

// ---------- flash-style causal attention ----------
// grid = BH(32) * Tq(128) blocks of ONE wave (32 threads). Each wave: 16 queries,
// streams keys in chunks of 32 with online softmax; QK^T: 4 wmma/chunk, PV: 4 wmma/chunk.
__global__ __launch_bounds__(32) void attn_kernel(
    const unsigned short* __restrict__ qg, const unsigned short* __restrict__ kg,
    const unsigned short* __restrict__ vtg, float* __restrict__ outg)
{
  __shared__ __align__(16) unsigned short Pl[16 * 32];  // P tile 16x32 bf16
  const int lane = threadIdx.x;
  const int hf = lane >> 4, l16 = lane & 15;
  const int qt = blockIdx.x & 127;
  const int bh = blockIdx.x >> 7;
  const int bb = bh >> 4, hh = bh & 15;
  const int qbase = qt * 16;

  const v8f zero8 = {0.f,0.f,0.f,0.f,0.f,0.f,0.f,0.f};

  // Q A-fragments (row = lane%16), loaded once
  Frag32 aq[2];
#pragma unroll
  for (int kf = 0; kf < 2; ++kf) {
    const unsigned short* p = qg + ((size_t)bh * 2048 + qbase + l16) * 64 + kf * 32 + hf * 8;
    aq[kf].u[0] = *(const v4u*)p;
    aq[kf].u[1] = *(const v4u*)(p + 16);
  }

  v8f acc[4] = {zero8, zero8, zero8, zero8};  // O tile 16x64
  float mrow[8], lrow[8];
#pragma unroll
  for (int j = 0; j < 8; ++j) { mrow[j] = -1e30f; lrow[j] = 0.f; }

  const int qe = qbase + 15;
  for (int kb = 0; kb <= qe; kb += 32) {
    // S = Q K^T for 32 keys -> two 16x16 f32 tiles
    v8f s[2] = {zero8, zero8};
#pragma unroll
    for (int t2 = 0; t2 < 2; ++t2) {
      int nidx = kb + t2 * 16 + l16;
      if (nidx > 2047) nidx = 2047;              // clamp (masked anyway)
#pragma unroll
      for (int kf = 0; kf < 2; ++kf) {
        Frag32 bk;
        const unsigned short* p = kg + ((size_t)bh * 2048 + nidx) * 64 + kf * 32 + hf * 8;
        bk.u[0] = *(const v4u*)p;
        bk.u[1] = *(const v4u*)(p + 16);
        s[t2] = wmma_bf16(aq[kf], bk, s[t2]);
      }
    }
    __syncthreads();  // prior-iteration Pl reads done before overwrite

    // online softmax update (C layout: row = j + 8*half, col = lane%16)
#pragma unroll
    for (int j = 0; j < 8; ++j) {
      int qrow = qbase + j + 8 * hf;
      int key0 = kb + l16;
      float s0 = (key0      <= qrow) ? s[0][j] * 0.125f : -1e30f;
      float s1 = (key0 + 16 <= qrow) ? s[1][j] * 0.125f : -1e30f;
      float mx = fmaxf(s0, s1);
#pragma unroll
      for (int mk = 1; mk < 16; mk <<= 1) mx = fmaxf(mx, __shfl_xor(mx, mk, 32));
      float mnew = fmaxf(mrow[j], mx);
      float al = (mrow[j] > -1e29f) ? __expf(mrow[j] - mnew) : 0.f;
      float p0 = (s0 > -1e29f) ? __expf(s0 - mnew) : 0.f;
      float p1 = (s1 > -1e29f) ? __expf(s1 - mnew) : 0.f;
      float ps = p0 + p1;
#pragma unroll
      for (int mk = 1; mk < 16; mk <<= 1) ps += __shfl_xor(ps, mk, 32);
      lrow[j] = lrow[j] * al + ps;
      mrow[j] = mnew;
#pragma unroll
      for (int nt = 0; nt < 4; ++nt) acc[nt][j] *= al;
      Pl[(j + 8 * hf) * 32 + l16]      = f2bf(p0);
      Pl[(j + 8 * hf) * 32 + 16 + l16] = f2bf(p1);
    }
    __syncthreads();

    // P A-fragment from LDS (re-layout C->A)
    Frag32 pa;
    {
      const unsigned short* p = &Pl[l16 * 32 + hf * 8];
      pa.u[0] = *(const v4u*)p;
      pa.u[1] = *(const v4u*)(p + 16);
    }
    // O += P @ V  (vT is [BH,64,T] so B-fragments are contiguous)
#pragma unroll
    for (int nt = 0; nt < 4; ++nt) {
      Frag32 bv;
      int kidx = kb + hf * 8;
      if (kidx > 2048 - 16) kidx = 2048 - 16;
      const unsigned short* p = vtg + ((size_t)bh * 64 + nt * 16 + l16) * 2048 + kidx;
      bv.u[0] = *(const v4u*)p;
      bv.u[1] = *(const v4u*)(p + 16);
      acc[nt] = wmma_bf16(pa, bv, acc[nt]);
    }
  }

  // finalize: O / l, scatter to [B,T,1024]
#pragma unroll
  for (int j = 0; j < 8; ++j) {
    float inv = lrow[j] > 0.f ? 1.f / lrow[j] : 0.f;
    int qrow = qbase + j + 8 * hf;
#pragma unroll
    for (int nt = 0; nt < 4; ++nt)
      outg[((size_t)bb * 2048 + qrow) * 1024 + hh * 64 + nt * 16 + l16] = acc[nt][j] * inv;
  }
}

// ---------- residual + layernorm (optionally also emit bf16 copy) ----------
__global__ __launch_bounds__(256) void ln_kernel(
    const float* __restrict__ a, const float* __restrict__ r,
    const float* __restrict__ g, const float* __restrict__ be,
    float* __restrict__ out, unsigned short* __restrict__ outbf)
{
  __shared__ float sm[16];
  const int row = blockIdx.x;
  const int t = threadIdx.x;
  float v[4], s = 0.f, sq = 0.f;
#pragma unroll
  for (int i = 0; i < 4; ++i) {
    int c = t + i * 256;
    float x = a[(size_t)row * 1024 + c] + r[(size_t)row * 1024 + c];
    v[i] = x; s += x; sq += x * x;
  }
#pragma unroll
  for (int mk = 1; mk < 32; mk <<= 1) { s += __shfl_xor(s, mk, 32); sq += __shfl_xor(sq, mk, 32); }
  int w = t >> 5;
  if ((t & 31) == 0) { sm[w] = s; sm[8 + w] = sq; }
  __syncthreads();
  if (t < 32) {
    float ss = (t < 8) ? sm[t] : 0.f;
    float qq = (t < 8) ? sm[8 + t] : 0.f;
#pragma unroll
    for (int mk = 1; mk < 8; mk <<= 1) { ss += __shfl_xor(ss, mk, 32); qq += __shfl_xor(qq, mk, 32); }
    if (t == 0) { sm[0] = ss; sm[8] = qq; }
  }
  __syncthreads();
  float mean = sm[0] * (1.f / 1024.f);
  float var  = sm[8] * (1.f / 1024.f) - mean * mean;
  float rstd = rsqrtf(var + 1e-5f);
#pragma unroll
  for (int i = 0; i < 4; ++i) {
    int c = t + i * 256;
    float y = (v[i] - mean) * rstd * g[c] + be[c];
    out[(size_t)row * 1024 + c] = y;
    if (outbf) outbf[(size_t)row * 1024 + c] = f2bf(y);
  }
}

// ---------- launch ----------
extern "C" void kernel_launch(void* const* d_in, const int* in_sizes, int n_in,
                              void* d_out, int out_size, void* d_ws, size_t ws_size,
                              hipStream_t stream) {
  const float* x    = (const float*)d_in[0];
  const float* Wqkv = (const float*)d_in[1];
  const float* bqkv = (const float*)d_in[2];
  const float* W1   = (const float*)d_in[3];
  const float* b1   = (const float*)d_in[4];
  const float* W2   = (const float*)d_in[5];
  const float* b2   = (const float*)d_in[6];
  const float* ln1g = (const float*)d_in[7];
  const float* ln1b = (const float*)d_in[8];
  const float* ln2g = (const float*)d_in[9];
  const float* ln2b = (const float*)d_in[10];

  char* ws = (char*)d_ws;
  size_t off = 0;
  auto alloc = [&](size_t bytes) -> void* {
    void* p = ws + off;
    off += (bytes + 255) & ~(size_t)255;
    return p;
  };
  const size_t M = 4096;  // B*T
  unsigned short* xb     = (unsigned short*)alloc(M * 1024 * 2);
  unsigned short* wqkvtb = (unsigned short*)alloc(1024ull * 3072 * 2);  // [3072][1024]
  unsigned short* w1tb   = (unsigned short*)alloc(1024ull * 4096 * 2);  // [4096][1024]
  unsigned short* w2tb   = (unsigned short*)alloc(4096ull * 1024 * 2);  // [1024][4096]
  unsigned short* qbuf   = (unsigned short*)alloc(M * 1024 * 2);
  unsigned short* kbuf   = (unsigned short*)alloc(M * 1024 * 2);
  unsigned short* vtbuf  = (unsigned short*)alloc(M * 1024 * 2);
  float*          attnb  = (float*)alloc(M * 1024 * 4);
  float*          hbuf   = (float*)alloc(M * 1024 * 4);
  unsigned short* hbb    = (unsigned short*)alloc(M * 1024 * 2);
  unsigned short* ff1b   = (unsigned short*)alloc(M * 4096 * 2);
  float*          ff2b   = (float*)alloc(M * 1024 * 4);

  cvt_bf16_kernel<<<512, 256, 0, stream>>>(x, xb, 4096 * 1024);
  cvt_bf16_t_kernel<<<512, 256, 0, stream>>>(Wqkv, wqkvtb, 1024, 3072);
  cvt_bf16_t_kernel<<<512, 256, 0, stream>>>(W1, w1tb, 1024, 4096);
  cvt_bf16_t_kernel<<<512, 256, 0, stream>>>(W2, w2tb, 4096, 1024);

  // QKV projection: [4096,1024] x [1024,3072] (B given transposed)
  gemm_kernel<0><<<dim3(3072 / 128, 4096 / 128), 256, 0, stream>>>(
      xb, wqkvtb, bqkv, qbuf, kbuf, vtbuf, 4096, 3072, 1024);

  // causal attention
  attn_kernel<<<32 * 128, 32, 0, stream>>>(qbuf, kbuf, vtbuf, attnb);

  // h = LN(x + attn), also bf16 copy for FFN
  ln_kernel<<<4096, 256, 0, stream>>>(x, attnb, ln1g, ln1b, hbuf, hbb);

  // FFN1: relu(h @ W1 + b1) -> bf16 [4096,4096]
  gemm_kernel<1><<<dim3(4096 / 128, 4096 / 128), 256, 0, stream>>>(
      hbb, w1tb, b1, ff1b, nullptr, nullptr, 4096, 4096, 1024);

  // FFN2: ff1 @ W2 + b2 -> f32 [4096,1024]
  gemm_kernel<2><<<dim3(1024 / 128, 4096 / 128), 256, 0, stream>>>(
      ff1b, w2tb, b2, ff2b, nullptr, nullptr, 4096, 1024, 4096);

  // out = LN(h + ff2)
  ln_kernel<<<4096, 256, 0, stream>>>(hbuf, ff2b, ln2g, ln2b, (float*)d_out, nullptr);
}
